// SLSTMEncoderZY_54022098649772
// MI455X (gfx1250) — compile-verified, matrix operand order checked by reading, not run on hardware
//
#include <hip/hip_runtime.h>
#include <hip/hip_bf16.h>
#include <math.h>

// Problem constants (B, S, H from the reference)
#define BD 8
#define SD 4096
#define HD 256
#define NLAYERS 7
#define EPSV 1e-3f
#define KBIG 1024           // 4*H : [hidden, h_before, h_after, emb]
#define NBIG 1536           // 6*H : 6 gates
#define BSH (BD*SD*HD)      // 8,388,608
#define BH  (BD*HD)         // 2,048
#define MROWS (BD*SD)       // 32,768

typedef __bf16 bf16_t;
typedef bf16_t v16bf __attribute__((ext_vector_type(16)));
typedef float  v8f   __attribute__((ext_vector_type(8)));

union Frag16 { v16bf v; unsigned int u[8]; };

__device__ __forceinline__ unsigned short f2bf(float f) {
  unsigned int u = __float_as_uint(f);
  u += 0x7FFFu + ((u >> 16) & 1u);      // round-to-nearest-even
  return (unsigned short)(u >> 16);
}
__device__ __forceinline__ float bf2f(unsigned short h) {
  return __uint_as_float(((unsigned int)h) << 16);
}
__device__ __forceinline__ float sigmoidf_(float x) { return 1.0f / (1.0f + __expf(-x)); }

__device__ __forceinline__ void loadB(const unsigned int* __restrict__ base, Frag16& f) {
  uint4 lo = *(const uint4*)(base);
  uint4 hi = *(const uint4*)(base + 4);
  f.u[0] = lo.x; f.u[1] = lo.y; f.u[2] = lo.z; f.u[3] = lo.w;
  f.u[4] = hi.x; f.u[5] = hi.y; f.u[6] = hi.z; f.u[7] = hi.w;
}

// ---------------------------------------------------------------------------
// Pack big combined weight matrix (4H x 6H) into per-lane WMMA B-fragment
// layout (bf16 pairs).  Fragment (kt, nt): 32 lanes x 8 dwords.
//   k = kt*32 + (v>>2)*16 + (lane>>4)*8 + (v&3)*2 ;  n = nt*16 + (lane&15)
// k-segments: 0:Wx[g]  1:Wh[g][:H]  2:Wh[g][H:]  3:Wi[g]     (g = n>>8)
// ---------------------------------------------------------------------------
__global__ void pack_big_kernel(const float* __restrict__ Wx, const float* __restrict__ Wh,
                                const float* __restrict__ Wi, unsigned int* __restrict__ pw) {
  int idx = blockIdx.x * 256 + threadIdx.x;     // total 32*96*32*8 = 786432
  int v = idx & 7;
  int t = idx >> 3;
  int lane = t & 31; t >>= 5;
  int nt = t % 96;
  int kt = t / 96;
  int k = (kt << 5) + ((v >> 2) << 4) + ((lane >> 4) << 3) + ((v & 3) << 1);
  int n = (nt << 4) + (lane & 15);
  int g = n >> 8, nn = n & 255;
  int seg = k >> 8, kr = k & 255;
  float f0, f1;
  if (seg == 0)      { f0 = Wx[(g*256 + kr)*256 + nn];       f1 = Wx[(g*256 + kr + 1)*256 + nn]; }
  else if (seg == 1) { f0 = Wh[(g*512 + kr)*256 + nn];       f1 = Wh[(g*512 + kr + 1)*256 + nn]; }
  else if (seg == 2) { f0 = Wh[(g*512 + 256 + kr)*256 + nn]; f1 = Wh[(g*512 + 257 + kr)*256 + nn]; }
  else               { f0 = Wi[(g*256 + kr)*256 + nn];       f1 = Wi[(g*256 + kr + 1)*256 + nn]; }
  pw[idx] = (unsigned int)f2bf(f0) | ((unsigned int)f2bf(f1) << 16);
}

// Pack gWh[2] (H x H) the same way (kt in [0,8), nt in [0,16))
__global__ void pack_gw_kernel(const float* __restrict__ gWh, unsigned int* __restrict__ pw) {
  int idx = blockIdx.x * 256 + threadIdx.x;     // total 8*16*32*8 = 32768
  int v = idx & 7;
  int t = idx >> 3;
  int lane = t & 31; t >>= 5;
  int nt = t & 15;
  int kt = t >> 4;
  int k = (kt << 5) + ((v >> 2) << 4) + ((lane >> 4) << 3) + ((v & 3) << 1);
  int n = (nt << 4) + (lane & 15);
  const float* W2 = gWh + 2 * 256 * 256;
  float f0 = W2[k * 256 + n];
  float f1 = W2[(k + 1) * 256 + n];
  pw[idx] = (unsigned int)f2bf(f0) | ((unsigned int)f2bf(f1) << 16);
}

// emb = inputs*m ; hidden = hidden0*m ; cell = cell0*m  (+ bf16 shadows)
__global__ void init_state_kernel(const float* __restrict__ inputs, const float* __restrict__ mask,
                                  const float* __restrict__ h0, const float* __restrict__ c0,
                                  float* __restrict__ emb, float* __restrict__ hid,
                                  float* __restrict__ cell, unsigned short* __restrict__ embbf,
                                  unsigned short* __restrict__ hbf) {
  int i = blockIdx.x * 256 + threadIdx.x;
  int bs = i >> 8;
  float m = mask[bs];
  float e = inputs[i] * m, h = h0[i] * m, c = c0[i] * m;
  emb[i] = e; hid[i] = h; cell[i] = c;
  embbf[i] = f2bf(e); hbf[i] = f2bf(h);
}

// ---- two-stage mean over S -------------------------------------------------
__global__ void mean_partial_kernel(const float* __restrict__ x, float* __restrict__ part) {
  int b = blockIdx.x, ch = blockIdx.y, k = threadIdx.x;       // grid (BD,16)
  const float* p = x + ((size_t)b * SD + ch * 256) * HD + k;
  float s = 0.f;
  for (int i = 0; i < 256; ++i) s += p[(size_t)i * HD];
  part[(b * 16 + ch) * HD + k] = s;
}
__global__ void mean_final_kernel(const float* __restrict__ part, float* __restrict__ out) {
  int b = blockIdx.x, k = threadIdx.x;
  float s = 0.f;
  for (int c = 0; c < 16; ++c) s += part[(b * 16 + c) * HD + k];
  out[b * HD + k] = s * (1.0f / SD);
}

// gx[g][b][k] = gh[b]·gWx[g][:,k] + (g<2 ? ch[b]·gWh[g][:,k] : 0)
__global__ void gate_matvec_kernel(const float* __restrict__ gh, const float* __restrict__ ch,
                                   const float* __restrict__ gWx, const float* __restrict__ gWh,
                                   float* __restrict__ gx) {
  int b = blockIdx.x, g = blockIdx.y, k = threadIdx.x;
  const float* wx = gWx + (size_t)g * HD * HD + k;
  const float* gv = gh + b * HD;
  float s = 0.f;
  for (int h = 0; h < HD; ++h) s += gv[h] * wx[(size_t)h * HD];
  if (g < 2) {
    const float* wh = gWh + (size_t)g * HD * HD + k;
    const float* cv = ch + b * HD;
    for (int h = 0; h < HD; ++h) s += cv[h] * wh[(size_t)h * HD];
  }
  gx[((size_t)g * BD + b) * HD + k] = s;
}

// gd (g=0), go (g=1) = sigmoid(LN(gx[g]) + gb[g])
__global__ void gdgo_kernel(const float* __restrict__ gx, const float* __restrict__ gln_g,
                            const float* __restrict__ gln_b, const float* __restrict__ gb,
                            float* __restrict__ gd, float* __restrict__ go) {
  __shared__ float red[HD];
  int b = blockIdx.x, g = blockIdx.y, k = threadIdx.x;
  float x = gx[((size_t)g * BD + b) * HD + k];
  red[k] = x; __syncthreads();
  for (int off = HD / 2; off > 0; off >>= 1) { if (k < off) red[k] += red[k + off]; __syncthreads(); }
  float mean = red[0] * (1.0f / HD); __syncthreads();
  float d = x - mean;
  red[k] = d * d; __syncthreads();
  for (int off = HD / 2; off > 0; off >>= 1) { if (k < off) red[k] += red[k + off]; __syncthreads(); }
  float rstd = rsqrtf(red[0] * (1.0f / HD) + EPSV);
  float y = d * rstd * gln_g[g * HD + k] + gln_b[g * HD + k] + gb[g * HD + k];
  (g == 0 ? gd : go)[b * HD + k] = sigmoidf_(y);
}

// ---------------------------------------------------------------------------
// gf1 = sigmoid( LN( gx2[b] + hidden@gWh2 ) + gb2 )   (WMMA, fused epilogue)
// Block: 64 rows x 256 cols; 8 waves = 4(M) x 2(N); wave: 1x8 tiles.
// Software-pipelined: one B fragment (and one A fragment) loaded ahead.
// ---------------------------------------------------------------------------
__global__ __launch_bounds__(256) void gemm_gf1_kernel(
    const unsigned short* __restrict__ hbf, const unsigned int* __restrict__ pw,
    const float* __restrict__ gx2, const float* __restrict__ gln_g2,
    const float* __restrict__ gln_b2, const float* __restrict__ gb2,
    float* __restrict__ gf1) {
  __shared__ __align__(16) unsigned char smem[64 * HD * 2 + 1024];
  unsigned int*   As    = (unsigned int*)smem;          // 64 x 128 dwords (bf16 pairs)
  unsigned short* gbuf  = (unsigned short*)smem;        // 64 x 256 bf16 (after GEMM)
  float*          stats = (float*)(smem + 64 * HD * 2);
  const int tid = threadIdx.x, wave = tid >> 5, lane = tid & 31;
  const int wave_m = wave >> 1, wave_n = wave & 1;
  const int row0 = blockIdx.x * 64, bb = row0 / SD;

  for (int i = tid; i < 64 * 128; i += 256)
    As[i] = ((const unsigned int*)hbf)[((size_t)row0 + (i >> 7)) * 128 + (i & 127)];
  __syncthreads();

  v8f acc[8];
  for (int t = 0; t < 8; ++t) for (int j = 0; j < 8; ++j) acc[t][j] = 0.f;
  const int m = lane & 15, half = lane >> 4;
  const int rowA = wave_m * 16 + m;
  const int nt0 = wave_n * 8;

  auto loadA = [&](int kt, Frag16& f) {
    #pragma unroll
    for (int v = 0; v < 8; ++v)
      f.u[v] = As[rowA * 128 + (kt << 4) + ((v >> 2) << 3) + (half << 2) + (v & 3)];
  };

  Frag16 a, bcur;
  loadA(0, a);
  loadB(pw + (((size_t)0 * 16 + nt0) * 32 + lane) * 8, bcur);
  #pragma unroll
  for (int step = 0; step < 64; ++step) {
    const int t = step & 7;
    Frag16 bnxt, anxt;
    if (step < 63) {
      const int s2 = step + 1, kt2 = s2 >> 3, t2 = s2 & 7;
      loadB(pw + (((size_t)kt2 * 16 + nt0 + t2) * 32 + lane) * 8, bnxt);
      if (t2 == 0) loadA(kt2, anxt);
    }
    acc[t] = __builtin_amdgcn_wmma_f32_16x16x32_bf16(false, a.v, false, bcur.v,
                                                     (short)0, acc[t], false, false);
    if (step < 63) { bcur = bnxt; if (((step + 1) & 7) == 0) a = anxt; }
  }
  __syncthreads();   // As dead; reuse as gbuf

  #pragma unroll
  for (int t = 0; t < 8; ++t) {
    int nb = (nt0 + t) * 16 + (lane & 15);
    float gv = gx2[bb * HD + nb];
    #pragma unroll
    for (int v = 0; v < 8; ++v) {
      int mo = wave_m * 16 + v + (half << 3);
      gbuf[mo * HD + nb] = f2bf(acc[t][v] + gv);
    }
  }
  __syncthreads();

  if (tid < 64) {
    const unsigned short* p = gbuf + tid * HD;
    float s = 0.f;
    for (int k = 0; k < HD; ++k) s += bf2f(p[k]);
    float mean = s * (1.0f / HD);
    float vv = 0.f;
    for (int k = 0; k < HD; ++k) { float d = bf2f(p[k]) - mean; vv += d * d; }
    stats[tid * 2] = mean;
    stats[tid * 2 + 1] = rsqrtf(vv * (1.0f / HD) + EPSV);
  }
  __syncthreads();

  for (int e = tid; e < 64 * HD; e += 256) {
    int r = e >> 8, k = e & 255;
    float x = bf2f(gbuf[r * HD + k]);
    float y = (x - stats[r * 2]) * stats[r * 2 + 1] * gln_g2[k] + gln_b2[k] + gb2[k];
    gf1[((size_t)(row0 + r)) * HD + k] = sigmoidf_(y);
  }
}

// wd_bias[b][g*H+k] = gh[b]·Wd[g][:,k]
__global__ void wd_bias_kernel(const float* __restrict__ gh, const float* __restrict__ Wd,
                               float* __restrict__ wdb) {
  int b = blockIdx.x, g = blockIdx.y, k = threadIdx.x;
  const float* w = Wd + (size_t)g * HD * HD + k;
  const float* gv = gh + b * HD;
  float s = 0.f;
  for (int h = 0; h < HD; ++h) s += gv[h] * w[(size_t)h * HD];
  wdb[(b * 6 + g) * HD + k] = s;
}

// ---- two-stage sequence softmax reduction ---------------------------------
__global__ void gc_partial_kernel(const float* __restrict__ gf1, const float* __restrict__ mask,
                                  const float* __restrict__ cell,
                                  float* __restrict__ pse, float* __restrict__ psec) {
  int b = blockIdx.x, ch = blockIdx.y, k = threadIdx.x;       // grid (BD,16)
  size_t base = ((size_t)b * SD + ch * 256) * HD + k;
  const float* mp = mask + (size_t)b * SD + ch * 256;
  float se = 0.f, sec = 0.f;
  for (int i = 0; i < 256; ++i) {
    float mv = mp[i];
    float e = (mv > 0.f) ? __expf(gf1[base + (size_t)i * HD]) : 0.f;   // masked -> exp(-inf)=0
    se += e;
    sec += e * cell[base + (size_t)i * HD];
  }
  pse[(b * 16 + ch) * HD + k] = se;
  psec[(b * 16 + ch) * HD + k] = sec;
}
__global__ void gc_final_kernel(const float* __restrict__ pse, const float* __restrict__ psec,
                                const float* __restrict__ gd, const float* __restrict__ go,
                                const float* __restrict__ gc_old,
                                float* __restrict__ gc_new, float* __restrict__ gh) {
  int b = blockIdx.x, k = threadIdx.x;
  float se = 0.f, sec = 0.f;
  for (int c = 0; c < 16; ++c) { se += pse[(b * 16 + c) * HD + k]; sec += psec[(b * 16 + c) * HD + k]; }
  float ed = __expf(gd[b * HD + k]);
  float gcn = (sec + ed * gc_old[b * HD + k]) / (se + ed);
  gc_new[b * HD + k] = gcn;
  gh[b * HD + k] = go[b * HD + k] * tanhf(gcn);
}

// ---------------------------------------------------------------------------
// Big GEMM (pure): pre[M=32768, N=1536] (bf16) = X[M,1024] @ Wcomb[1024,1536]
// Block tile 128x128, 8 waves = 4(M) x 2(N); wave: 2x4 tiles of 16x16.
// 4-deep LDS pipeline: iteration kt computes from buf[kt&3] and stages chunk
// kt+2 at the END of the iteration, so B-fragment waits never transitively
// wait on staging loads (LOADcnt decrements in order).
// ---------------------------------------------------------------------------
__device__ __forceinline__ void stage_chunk(int kt, unsigned int* dst, int tid,
                                            const unsigned short* __restrict__ hbf,
                                            const unsigned short* __restrict__ embbf,
                                            int bb, int s0) {
  int seg = kt >> 3;                 // 32-chunk segment (uniform per iteration)
  int kk0 = (kt & 7) << 5;
  int r = tid >> 1, halfr = tid & 1; // thread loads 32B of one row
  int ss = s0 + r + ((seg == 1) ? -1 : (seg == 2 ? 1 : 0));
  const unsigned short* src = (seg == 3) ? embbf : hbf;
  uint4 lo = make_uint4(0, 0, 0, 0), hi = make_uint4(0, 0, 0, 0);
  if (ss >= 0 && ss < SD) {
    const uint4* p = (const uint4*)(src + ((size_t)(bb * SD + ss)) * HD + kk0 + halfr * 16);
    lo = p[0]; hi = p[1];
  }
  uint4* d = (uint4*)(dst + r * 16 + halfr * 8);
  d[0] = lo; d[1] = hi;
}

__global__ __launch_bounds__(256) void gemm_pre_kernel(
    const unsigned short* __restrict__ hbf, const unsigned short* __restrict__ embbf,
    const unsigned int* __restrict__ pw, unsigned short* __restrict__ pre) {
  __shared__ __align__(16) unsigned int As[4][128 * 16];   // 4 x 8KB chunks
  const int tid = threadIdx.x, wave = tid >> 5, lane = tid & 31;
  const int wave_m = wave >> 1, wave_n = wave & 1;
  const int row0 = blockIdx.x * 128, bb = row0 / SD, s0 = row0 % SD;
  const int ntbase = blockIdx.y * 8 + wave_n * 4;          // of 96 n-tiles

  stage_chunk(0, As[0], tid, hbf, embbf, bb, s0);
  stage_chunk(1, As[1], tid, hbf, embbf, bb, s0);
  __syncthreads();

  v8f acc[2][4];
  for (int tm = 0; tm < 2; ++tm) for (int tn = 0; tn < 4; ++tn)
    for (int j = 0; j < 8; ++j) acc[tm][tn][j] = 0.f;
  const int m = lane & 15, half = lane >> 4;

  for (int kt = 0; kt < 32; ++kt) {
    const unsigned int* Ab = As[kt & 3];

    Frag16 afr[2];
    #pragma unroll
    for (int tm = 0; tm < 2; ++tm) {
      int row = wave_m * 32 + tm * 16 + m;
      #pragma unroll
      for (int v = 0; v < 8; ++v)
        afr[tm].u[v] = Ab[row * 16 + ((v >> 2) << 3) + (half << 2) + (v & 3)];
    }
    const unsigned int* bbase = pw + (((size_t)kt * 96 + ntbase) * 32 + lane) * 8;
    if (kt + 1 < 32)
      __builtin_prefetch(pw + (((size_t)(kt + 1) * 96 + ntbase) * 32 + lane) * 8, 0, 0);

    Frag16 bcur;
    loadB(bbase, bcur);
    #pragma unroll
    for (int tn = 0; tn < 4; ++tn) {
      Frag16 bnxt;
      if (tn < 3) loadB(bbase + (tn + 1) * 256, bnxt);
      #pragma unroll
      for (int tm = 0; tm < 2; ++tm)
        acc[tm][tn] = __builtin_amdgcn_wmma_f32_16x16x32_bf16(false, afr[tm].v, false, bcur.v,
                                                              (short)0, acc[tm][tn], false, false);
      if (tn < 3) bcur = bnxt;
    }

    if (kt + 2 < 32) stage_chunk(kt + 2, As[(kt + 2) & 3], tid, hbf, embbf, bb, s0);
    __syncthreads();
  }

  #pragma unroll
  for (int tm = 0; tm < 2; ++tm) {
    #pragma unroll
    for (int tn = 0; tn < 4; ++tn) {
      int nb = (ntbase + tn) * 16 + (lane & 15);
      #pragma unroll
      for (int v = 0; v < 8; ++v) {
        int row = row0 + wave_m * 32 + tm * 16 + v + (half << 3);
        pre[(size_t)row * NBIG + nb] = f2bf(acc[tm][tn][v]);
      }
    }
  }
}

// ---------------------------------------------------------------------------
// Epilogue: +Wd bias, LN per gate, sigmoid, gate softmax, state update.
// Block: 16 rows x 1536 gate-columns (48KB bf16 in LDS).
// ---------------------------------------------------------------------------
__global__ __launch_bounds__(256) void epilogue_kernel(
    const unsigned short* __restrict__ pre, const float* __restrict__ wdb,
    const float* __restrict__ lng, const float* __restrict__ lnb, const float* __restrict__ bias,
    const float* __restrict__ cell, const float* __restrict__ emb, const float* __restrict__ hid,
    const float* __restrict__ tgc, const float* __restrict__ mask,
    float* __restrict__ h_out, float* __restrict__ c_out, unsigned short* __restrict__ hbf_out) {
  __shared__ __align__(16) unsigned char smem[16 * NBIG * 2 + 1024];
  unsigned short* gbuf  = (unsigned short*)smem;        // 16 x 1536 bf16
  float*          stats = (float*)(smem + 16 * NBIG * 2);
  const int tid = threadIdx.x;
  const int row0 = blockIdx.x * 16, bb = row0 / SD, s0 = row0 % SD;

  // load pre tile, add Wd bias
  for (int i = tid; i < 16 * 768; i += 256) {           // dword pairs
    int r = i / 768, d = i % 768;
    unsigned int u = ((const unsigned int*)pre)[((size_t)(row0 + r)) * 768 + d];
    int n = d * 2;
    float x0 = bf2f((unsigned short)(u & 0xffffu)) + wdb[bb * NBIG + n];
    float x1 = bf2f((unsigned short)(u >> 16)) + wdb[bb * NBIG + n + 1];
    ((unsigned int*)gbuf)[r * 768 + d] = (unsigned int)f2bf(x0) | ((unsigned int)f2bf(x1) << 16);
  }
  __syncthreads();

  for (int idx = tid; idx < 96; idx += 256) {           // LN stats per (row, gate)
    int r = idx / 6, g = idx % 6;
    const unsigned short* p = gbuf + r * NBIG + g * HD;
    float s = 0.f;
    for (int k = 0; k < HD; ++k) s += bf2f(p[k]);
    float mean = s * (1.0f / HD);
    float vv = 0.f;
    for (int k = 0; k < HD; ++k) { float d = bf2f(p[k]) - mean; vv += d * d; }
    stats[idx * 2] = mean;
    stats[idx * 2 + 1] = rsqrtf(vv * (1.0f / HD) + EPSV);
  }
  __syncthreads();

  for (int e = tid; e < 16 * HD; e += 256) {
    int r = e >> 8, k = e & 255;
    int s = s0 + r;
    size_t base = ((size_t)(bb * SD + s)) * HD + k;
    float gate[6];
    #pragma unroll
    for (int g = 0; g < 6; ++g) {
      float x = bf2f(gbuf[r * NBIG + g * HD + k]);
      float mean = stats[(r * 6 + g) * 2], rstd = stats[(r * 6 + g) * 2 + 1];
      float y = (x - mean) * rstd * lng[g * HD + k] + lnb[g * HD + k] + bias[g * HD + k];
      gate[g] = sigmoidf_(y);
    }
    float e0 = __expf(gate[0]), e1 = __expf(gate[1]), e2 = __expf(gate[2]);
    float e3 = __expf(gate[3]), e4 = __expf(gate[4]);
    float inv = 1.0f / (e0 + e1 + e2 + e3 + e4);
    float cb = (s > 0)      ? cell[base - HD] : 0.f;
    float ca = (s < SD - 1) ? cell[base + HD] : 0.f;
    float cc = cell[base], ee = emb[base], hh = hid[base];
    float tg = tgc[bb * HD + k];
    float mv = mask[bb * SD + s];
    // f1,f2,f3,f4,i_t = softmax(gates[:5]); o_t = gates[5]
    float ct = (cb * e0 + ca * e1 + cc * e4 + ee * e2 + tg * e3) * inv;
    float ht = gate[5] * (ct + hh);
    float hn = ht * mv, cn = ct * mv;
    h_out[base] = hn;
    c_out[base] = cn;
    hbf_out[base] = f2bf(hn);
  }
}

__global__ void finalize_kernel(const float* __restrict__ hid, const float* __restrict__ gh,
                                float* __restrict__ out) {
  int i = blockIdx.x * 256 + threadIdx.x;
  if (i < BSH) out[i] = hid[i];
  else if (i < BSH + BH) out[i] = gh[i - BSH];
}

// ---------------------------------------------------------------------------
extern "C" void kernel_launch(void* const* d_in, const int* in_sizes, int n_in,
                              void* d_out, int out_size, void* d_ws, size_t ws_size,
                              hipStream_t stream) {
  (void)in_sizes; (void)n_in; (void)out_size; (void)ws_size;
  const float* inputs = (const float*)d_in[0];
  const float* mask   = (const float*)d_in[1];
  const float* h0     = (const float*)d_in[2];
  const float* c0     = (const float*)d_in[3];
  const float* Wx     = (const float*)d_in[4];
  const float* Wh     = (const float*)d_in[5];
  const float* Wi     = (const float*)d_in[6];
  const float* Wd     = (const float*)d_in[7];
  const float* bvec   = (const float*)d_in[8];
  const float* ln_g   = (const float*)d_in[9];
  const float* ln_b   = (const float*)d_in[10];
  const float* gWx    = (const float*)d_in[11];
  const float* gWh    = (const float*)d_in[12];
  const float* gb     = (const float*)d_in[13];
  const float* gln_g  = (const float*)d_in[14];
  const float* gln_b  = (const float*)d_in[15];

  char* ws = (char*)d_ws;
  size_t off = 0;
  auto alloc = [&](size_t bytes) -> char* {
    char* p = ws + off; off += (bytes + 255) & ~(size_t)255; return p;
  };
  float* hA  = (float*)alloc((size_t)BSH * 4);
  float* hB  = (float*)alloc((size_t)BSH * 4);
  float* cA  = (float*)alloc((size_t)BSH * 4);
  float* cB  = (float*)alloc((size_t)BSH * 4);
  float* emb = (float*)alloc((size_t)BSH * 4);
  float* gf1 = (float*)alloc((size_t)BSH * 4);
  unsigned short* hbfA  = (unsigned short*)alloc((size_t)BSH * 2);
  unsigned short* hbfB  = (unsigned short*)alloc((size_t)BSH * 2);
  unsigned short* embbf = (unsigned short*)alloc((size_t)BSH * 2);
  unsigned short* pre   = (unsigned short*)alloc((size_t)MROWS * NBIG * 2);   // 100.7MB
  unsigned int* pw  = (unsigned int*)alloc(786432 * 4);
  unsigned int* pgw = (unsigned int*)alloc(32768 * 4);
  float* gx   = (float*)alloc(3 * BH * 4);
  float* gd   = (float*)alloc(BH * 4);
  float* go   = (float*)alloc(BH * 4);
  float* gcA  = (float*)alloc(BH * 4);
  float* gcB  = (float*)alloc(BH * 4);
  float* gh   = (float*)alloc(BH * 4);
  float* chm  = (float*)alloc(BH * 4);
  float* wdb  = (float*)alloc(BD * NBIG * 4);
  float* part = (float*)alloc(16 * BH * 4);
  float* pse  = (float*)alloc(16 * BH * 4);
  float* psec = (float*)alloc(16 * BH * 4);

  pack_big_kernel<<<3072, 256, 0, stream>>>(Wx, Wh, Wi, pw);
  pack_gw_kernel<<<128, 256, 0, stream>>>(gWh, pgw);
  init_state_kernel<<<BSH / 256, 256, 0, stream>>>(inputs, mask, h0, c0, emb, hA, cA, embbf, hbfA);
  mean_partial_kernel<<<dim3(BD, 16), HD, 0, stream>>>(hA, part);
  mean_final_kernel<<<BD, HD, 0, stream>>>(part, gh);
  mean_partial_kernel<<<dim3(BD, 16), HD, 0, stream>>>(cA, part);
  mean_final_kernel<<<BD, HD, 0, stream>>>(part, gcA);

  float* h_cur = hA; float* h_nxt = hB;
  float* c_cur = cA; float* c_nxt = cB;
  unsigned short* hbf_cur = hbfA; unsigned short* hbf_nxt = hbfB;
  float* gc_cur = gcA; float* gc_nxt = gcB;

  for (int layer = 0; layer < NLAYERS; ++layer) {
    mean_partial_kernel<<<dim3(BD, 16), HD, 0, stream>>>(h_cur, part);
    mean_final_kernel<<<BD, HD, 0, stream>>>(part, chm);
    gate_matvec_kernel<<<dim3(BD, 3), HD, 0, stream>>>(gh, chm, gWx, gWh, gx);
    gdgo_kernel<<<dim3(BD, 2), HD, 0, stream>>>(gx, gln_g, gln_b, gb, gd, go);
    gemm_gf1_kernel<<<MROWS / 64, 256, 0, stream>>>(hbf_cur, pgw, gx + 2 * BH,
                                                    gln_g + 2 * HD, gln_b + 2 * HD,
                                                    gb + 2 * HD, gf1);
    wd_bias_kernel<<<dim3(BD, 6), HD, 0, stream>>>(gh, Wd, wdb);   // uses gh_old
    gc_partial_kernel<<<dim3(BD, 16), HD, 0, stream>>>(gf1, mask, c_cur, pse, psec);
    gc_final_kernel<<<BD, HD, 0, stream>>>(pse, psec, gd, go, gc_cur, gc_nxt, gh);
    gemm_pre_kernel<<<dim3(MROWS / 128, NBIG / 128), 256, 0, stream>>>(hbf_cur, embbf, pw, pre);
    epilogue_kernel<<<MROWS / 16, 256, 0, stream>>>(pre, wdb, ln_g, ln_b, bvec,
                                                    c_cur, emb, h_cur,
                                                    gc_cur /* tgc = old gc */, mask,
                                                    h_nxt, c_nxt, hbf_nxt);
    float* tf;
    tf = h_cur; h_cur = h_nxt; h_nxt = tf;
    tf = c_cur; c_cur = c_nxt; c_nxt = tf;
    unsigned short* ts = hbf_cur; hbf_cur = hbf_nxt; hbf_nxt = ts;
    tf = gc_cur; gc_cur = gc_nxt; gc_nxt = tf;
  }

  finalize_kernel<<<(BSH + BH + 255) / 256, 256, 0, stream>>>(h_cur, gh, (float*)d_out);
}
